// _NepaSelfAttention_84447646974276
// MI455X (gfx1250) — compile-verified, matrix operand order checked by reading, not run on hardware
//
#include <hip/hip_runtime.h>
#include <hip/hip_bf16.h>

// ---------------- CDNA5 WMMA types ----------------
typedef __attribute__((ext_vector_type(16))) __bf16 v16bf;
typedef __attribute__((ext_vector_type(8)))  float  v8f;
typedef __attribute__((ext_vector_type(4)))  unsigned int u32x4;
typedef __attribute__((ext_vector_type(8)))  int i32x8;
typedef __attribute__((ext_vector_type(4)))  int i32x4;

#define NEPA_HAS_TDM __has_builtin(__builtin_amdgcn_tensor_load_to_lds)

union FragBF {
    v16bf v;
    unsigned short u[16];
    uint4 q[2];
};

#define B_  2
#define T_  2048
#define D_  1024
#define H_  16
#define DH_ 64
#define SCALE_ 0.125f
#define MROWS_ 4096   // B_*T_

static __device__ __forceinline__ unsigned short f2bf(float f) {
    unsigned int u = __float_as_uint(f);
    unsigned int r = (u + 0x7FFFu + ((u >> 16) & 1u)) >> 16;   // RNE
    return (unsigned short)r;
}

static __device__ __forceinline__ v8f wmma_bf16(v16bf a, v16bf b, v8f c) {
    // D = A(16x32 bf16) x B(32x16 bf16) + C(16x16 f32)
    return __builtin_amdgcn_wmma_f32_16x16x32_bf16(
        /*neg_a=*/false, a, /*neg_b=*/false, b,
        /*c_mod=*/(short)0, c, /*reuse_a=*/false, /*reuse_b=*/false);
}

// Load an A-style 16x32 bf16 fragment from row-major memory (rows = M or N,
// contiguous K). Per ISA 16-bit A layout: lane<16 -> K groups {0-7,16-23},
// lane>=16 -> {8-15,24-31}; both groups are contiguous 16B runs.
static __device__ __forceinline__ v16bf load_frag_rows(const unsigned short* __restrict__ base,
                                                       int ld, int row0, int k0) {
    int lane = threadIdx.x & 31;
    int m  = lane & 15;
    int kh = (lane >> 4) & 1;
    const unsigned short* p = base + (size_t)(row0 + m) * (size_t)ld + k0 + kh * 8;
    FragBF f;
    f.q[0] = *(const uint4*)(p);        // K = kh*8 + 0..7
    f.q[1] = *(const uint4*)(p + 16);   // K = 16 + kh*8 + 0..7
    return f.v;
}

// Same mapping, but reading from an LDS row-major tile (used for P fragment).
static __device__ __forceinline__ v16bf load_frag_lds(const unsigned short* p0, int ld) {
    int lane = threadIdx.x & 31;
    int m  = lane & 15;
    int kh = (lane >> 4) & 1;
    const unsigned short* p = p0 + m * ld + kh * 8;
    FragBF f;
#pragma unroll
    for (int h = 0; h < 8; ++h)  f.u[h]     = p[h];
#pragma unroll
    for (int h = 0; h < 8; ++h)  f.u[8 + h] = p[16 + h];
    return f.v;
}

// B fragment gather from an LDS tile stored [K][N] row-major over N:
// lane owns column n, K runs per the A-layout K(h) mapping.
static __device__ __forceinline__ v16bf load_fragB_col(const unsigned short* vt,
                                                       int ldn, int ncol0) {
    int lane = threadIdx.x & 31;
    int n  = ncol0 + (lane & 15);
    int kh = ((lane >> 4) & 1) * 8;
    FragBF f;
#pragma unroll
    for (int h = 0; h < 16; ++h) {
        int k = (h & 7) + ((h & 8) << 1) + kh;
        f.u[h] = vt[k * ldn + n];
    }
    return f.v;
}

// ---------------- kernel 1: f32 -> bf16 convert ----------------
__global__ void nepa_cvt_bf16(const float* __restrict__ src,
                              unsigned short* __restrict__ dst, int n) {
    for (int i = blockIdx.x * blockDim.x + threadIdx.x; i < n; i += gridDim.x * blockDim.x)
        dst[i] = f2bf(src[i]);
}

// ---------------- kernel 2: WMMA GEMM  Y = A * W^T + bias ----------------
// Wave computes a 32(m) x 64(n) C tile: 2 A-frags + 4 B-frags -> 8 WMMAs / k-step.
// A: [4096 x 1024] bf16 row-major.  W: [1024 x 1024] bf16 row-major (e,d),
// so B[k][n] = W[n][k] -> lane-per-row loads of W rows.
// mode 0: f32 out, head-major [b,h,t,dh]
// mode 1: bf16 out, head-major (for V)
// mode 2: f32 out, [b,t,e] flat (final projection -> d_out)
__global__ void nepa_gemm_wmma(const unsigned short* __restrict__ A,
                               const unsigned short* __restrict__ W,
                               const float* __restrict__ bias,
                               void* __restrict__ dst, int mode) {
    int wave = threadIdx.x >> 5;
    int m0 = blockIdx.x * 128 + wave * 32;
    int n0 = blockIdx.y * 64;

    v8f acc[2][4] = {};
    for (int k = 0; k < D_; k += 32) {
        if (k + 32 < D_)
            __builtin_prefetch((const void*)(A + (size_t)m0 * D_ + k + 32), 0, 0);
        v16bf a0 = load_frag_rows(A, D_, m0,      k);
        v16bf a1 = load_frag_rows(A, D_, m0 + 16, k);
#pragma unroll
        for (int j = 0; j < 4; ++j) {
            v16bf b = load_frag_rows(W, D_, n0 + j * 16, k);
            acc[0][j] = wmma_bf16(a0, b, acc[0][j]);
            acc[1][j] = wmma_bf16(a1, b, acc[1][j]);
        }
    }

    int lane  = threadIdx.x & 31;
    int col_l = lane & 15;
    int rhi   = ((lane >> 4) & 1) * 8;
#pragma unroll
    for (int mi = 0; mi < 2; ++mi) {
#pragma unroll
        for (int j = 0; j < 4; ++j) {
            int col = n0 + j * 16 + col_l;
            float bv = bias[col];
#pragma unroll
            for (int i = 0; i < 8; ++i) {
                int r = m0 + mi * 16 + i + rhi;
                float val = acc[mi][j][i] + bv;
                if (mode == 2) {
                    ((float*)dst)[(size_t)r * D_ + col] = val;
                } else {
                    size_t idx = (((size_t)(r >> 11) * H_ + (col >> 6)) * T_ + (r & (T_ - 1))) * DH_
                                 + (col & (DH_ - 1));
                    if (mode == 0) ((float*)dst)[idx] = val;
                    else           ((unsigned short*)dst)[idx] = f2bf(val);
                }
            }
        }
    }
}

// ---------------- kernel 3: per-head LayerNorm + RoPE -> bf16 ----------------
// One wave per (b,h,t) row of 64. Lane owns elements {lane, lane+32}, which are
// exactly the rotate-half partners -> RoPE is lane-local.
__global__ void nepa_ln_rope(const float* __restrict__ src,
                             const float* __restrict__ w,
                             unsigned short* __restrict__ dst) {
    int row  = blockIdx.x * 8 + (threadIdx.x >> 5);
    int lane = threadIdx.x & 31;
    float x0 = src[(size_t)row * DH_ + lane];
    float x1 = src[(size_t)row * DH_ + 32 + lane];

    float s = x0 + x1;
#pragma unroll
    for (int o = 16; o >= 1; o >>= 1) s += __shfl_xor(s, o, 32);
    float mean = s * (1.0f / DH_);
    float d0 = x0 - mean, d1 = x1 - mean;
    float vs = d0 * d0 + d1 * d1;
#pragma unroll
    for (int o = 16; o >= 1; o >>= 1) vs += __shfl_xor(vs, o, 32);
    float rinv = rsqrtf(vs * (1.0f / DH_) + 1e-6f);

    float y0 = d0 * rinv * w[lane];
    float y1 = d1 * rinv * w[lane + 32];

    int t = row & (T_ - 1);
    float o0 = y0, o1 = y1;
    if (t >= 1) {  // ROPE_PREFIX = 1
        float mrow  = (float)(t - 1);
        float coord = 2.0f * ((mrow + 0.5f) / (float)(T_ - 1)) - 1.0f;
        // inv_freq[f] = 100^(-f/16), f = j mod 16 (same for j and j+32)
        float freq = __expf(-(float)(lane & 15) * (4.60517019f / 16.0f));
        float ang  = 6.28318531f * coord * freq;
        float c = __cosf(ang), sn = __sinf(ang);
        o0 = y0 * c - y1 * sn;   // rot[j]   = -x[j+32], j < 32
        o1 = y1 * c + y0 * sn;   // rot[j+32] = x[j]
    }
    dst[(size_t)row * DH_ + lane]      = f2bf(o0);
    dst[(size_t)row * DH_ + 32 + lane] = f2bf(o1);
}

// ---------------- kernel 4: flash attention (WMMA + TDM, online softmax) ----------------
// grid (T/64, B*H), block 128 (4 waves). Wave owns 16 Q rows; KV advances by 32.
// The shared 32x64 bf16 V tile (contiguous 4KB) is DMA'd to LDS by the Tensor
// Data Mover (wave 0 issues, TENSORcnt-tracked), overlapping P-tile stores.
__global__ void nepa_flash_attn(const unsigned short* __restrict__ Qb,
                                const unsigned short* __restrict__ Kb,
                                const unsigned short* __restrict__ Vb,
                                const float* __restrict__ mask,
                                unsigned short* __restrict__ ctx) {
    __shared__ unsigned short ldsV[32 * 64];       // shared KV value tile
    __shared__ unsigned short ldsP[4][16 * 32];    // per-wave P tile

    int tid  = threadIdx.x;
    int wave = tid >> 5;
    int lane = tid & 31;
    int bh   = blockIdx.y;
    int q0   = blockIdx.x * 64 + wave * 16;
    size_t base = (size_t)bh * T_ * DH_;

    v16bf aQ0 = load_frag_rows(Qb + base, DH_, q0, 0);
    v16bf aQ1 = load_frag_rows(Qb + base, DH_, q0, 32);

    float mrow[8], lrow[8];
#pragma unroll
    for (int j = 0; j < 8; ++j) { mrow[j] = -1e30f; lrow[j] = 0.0f; }
    v8f acc[4] = {};

    int coll = lane & 15;
    int rhi  = ((lane >> 4) & 1) * 8;

#if NEPA_HAS_TDM
    unsigned lds_vbase =
        (unsigned)(unsigned long long)(__attribute__((address_space(3))) unsigned short*)ldsV;
#endif

    for (int kt = 0; kt < T_; kt += 32) {
        // ---- S = Q K^T for two 16-wide KV tiles (4 WMMAs) ----
        v8f s0 = {}, s1 = {};
        s0 = wmma_bf16(aQ0, load_frag_rows(Kb + base, DH_, kt, 0),       s0);
        s0 = wmma_bf16(aQ1, load_frag_rows(Kb + base, DH_, kt, 32),      s0);
        s1 = wmma_bf16(aQ0, load_frag_rows(Kb + base, DH_, kt + 16, 0),  s1);
        s1 = wmma_bf16(aQ1, load_frag_rows(Kb + base, DH_, kt + 16, 32), s1);

        // ---- online softmax update ----
        float p0v[8], p1v[8];
#pragma unroll
        for (int j = 0; j < 8; ++j) {
            int rg = q0 + j + rhi;
            float v0 = s0[j] * SCALE_ + mask[(size_t)rg * T_ + kt + coll];
            float v1 = s1[j] * SCALE_ + mask[(size_t)rg * T_ + kt + 16 + coll];
            float tm = fmaxf(v0, v1);
#pragma unroll
            for (int o = 8; o >= 1; o >>= 1) tm = fmaxf(tm, __shfl_xor(tm, o, 32));
            float mn    = fmaxf(mrow[j], tm);
            float alpha = __expf(mrow[j] - mn);
            mrow[j] = mn;
            float p0 = __expf(v0 - mn), p1 = __expf(v1 - mn);
            float rs = p0 + p1;
#pragma unroll
            for (int o = 8; o >= 1; o >>= 1) rs += __shfl_xor(rs, o, 32);
            lrow[j] = lrow[j] * alpha + rs;
            acc[0][j] *= alpha; acc[1][j] *= alpha;
            acc[2][j] *= alpha; acc[3][j] *= alpha;
            p0v[j] = p0; p1v[j] = p1;
        }

        __syncthreads();   // previous iteration's LDS reads complete

        // ---- stage V tile (TDM DMA if available, else cooperative copy) ----
#if NEPA_HAS_TDM
        if (wave == 0) {
            unsigned long long gaddr =
                (unsigned long long)(const void*)(Vb + base + (size_t)kt * DH_);
            u32x4 g0;
            g0[0] = 1u;                                        // count=1, load descriptor
            g0[1] = lds_vbase;                                 // lds_addr
            g0[2] = (unsigned)(gaddr & 0xFFFFFFFFu);           // global_addr[31:0]
            g0[3] = (unsigned)((gaddr >> 32) & 0x1FFFFFFu)     // global_addr[56:32]
                    | (2u << 30);                              // type = 2 (image)
            i32x8 g1;
            g1[0] = 0x00010000;        // workgroup_mask=0, data_size=1 (2 bytes)
            g1[1] = 0;                 // no atomic barrier; tensor_dim0[15:0]=0
            g1[2] = 0x00010002;        // tensor_dim0 = 0x20000 (131072); tensor_dim1 = 1
            g1[3] = (int)(2048u << 16);// tensor_dim1 hi = 0; tile_dim0 = 2048 elements
            g1[4] = 1;                 // tile_dim1 = 1, tile_dim2 = 0
            g1[5] = 2048;              // tensor_dim0_stride = 2048
            g1[6] = 0;
            g1[7] = 0;
            i32x4 z4 = {};
#if __clang_major__ >= 23
            i32x8 z8 = {};
            __builtin_amdgcn_tensor_load_to_lds(g0, g1, z4, z4, z8, 0);
#else
            __builtin_amdgcn_tensor_load_to_lds(g0, g1, z4, z4, 0);
#endif
        }
#endif

        // stage P (per-wave) in LDS while the TDM transfer is in flight
#pragma unroll
        for (int j = 0; j < 8; ++j) {
            ldsP[wave][(j + rhi) * 32 + coll]      = f2bf(p0v[j]);
            ldsP[wave][(j + rhi) * 32 + 16 + coll] = f2bf(p1v[j]);
        }

#if NEPA_HAS_TDM
        if (wave == 0) {
#if __has_builtin(__builtin_amdgcn_s_wait_tensorcnt)
            __builtin_amdgcn_s_wait_tensorcnt(0);
#else
            asm volatile("s_wait_tensorcnt 0x0" ::: "memory");
#endif
        }
#else
        {
            const uint4* vsrc = (const uint4*)(Vb + base + (size_t)kt * DH_);
            uint4* vdst = (uint4*)ldsV;
            vdst[tid]       = vsrc[tid];
            vdst[tid + 128] = vsrc[tid + 128];
        }
#endif

        __syncthreads();   // LDS tiles visible

        // ---- O += P(16x32) * V(32x64)  (4 WMMAs) ----
        v16bf aP = load_frag_lds(&ldsP[wave][0], 32);
#pragma unroll
        for (int j = 0; j < 4; ++j) {
            v16bf bV = load_fragB_col(ldsV, DH_, j * 16);
            acc[j] = wmma_bf16(aP, bV, acc[j]);
        }
    }

    // ---- normalize and write ctx[b, t, h*64 + n] as bf16 ----
    int b = bh >> 4, h = bh & 15;
#pragma unroll
    for (int j = 0; j < 8; ++j) {
        float inv = 1.0f / lrow[j];
        int trow = q0 + j + rhi;
#pragma unroll
        for (int nt = 0; nt < 4; ++nt) {
            int e = h * DH_ + nt * 16 + coll;
            ctx[((size_t)b * T_ + trow) * D_ + e] = f2bf(acc[nt][j] * inv);
        }
    }
}

// ---------------- host launcher ----------------
extern "C" void kernel_launch(void* const* d_in, const int* in_sizes, int n_in,
                              void* d_out, int out_size, void* d_ws, size_t ws_size,
                              hipStream_t stream) {
    const float* x    = (const float*)d_in[0];
    const float* mask = (const float*)d_in[1];
    const float* Wq   = (const float*)d_in[2];
    const float* bq   = (const float*)d_in[3];
    const float* Wk   = (const float*)d_in[4];
    const float* bk   = (const float*)d_in[5];
    const float* Wv   = (const float*)d_in[6];
    const float* bv   = (const float*)d_in[7];
    const float* Wp   = (const float*)d_in[8];
    const float* bp   = (const float*)d_in[9];
    const float* qn   = (const float*)d_in[10];
    const float* kn   = (const float*)d_in[11];

    const size_t MB = 1u << 20;
    char* ws = (char*)d_ws;
    unsigned short* xb  = (unsigned short*)(ws + 0 * MB);   // 8 MB
    unsigned short* wqb = (unsigned short*)(ws + 8 * MB);   // 2 MB
    unsigned short* wkb = (unsigned short*)(ws + 10 * MB);
    unsigned short* wvb = (unsigned short*)(ws + 12 * MB);
    unsigned short* wpb = (unsigned short*)(ws + 14 * MB);
    float*          qf  = (float*)(ws + 16 * MB);           // 16 MB
    float*          kf  = (float*)(ws + 32 * MB);           // 16 MB
    unsigned short* qbb = (unsigned short*)(ws + 48 * MB);  // 8 MB
    unsigned short* kbb = (unsigned short*)(ws + 56 * MB);  // 8 MB
    unsigned short* vbb = (unsigned short*)(ws + 64 * MB);  // 8 MB
    unsigned short* ctx = (unsigned short*)(ws + 72 * MB);  // 8 MB

    // 1) convert activations + weights to bf16
    nepa_cvt_bf16<<<4096, 256, 0, stream>>>(x,  xb,  B_ * T_ * D_);
    nepa_cvt_bf16<<<1024, 256, 0, stream>>>(Wq, wqb, D_ * D_);
    nepa_cvt_bf16<<<1024, 256, 0, stream>>>(Wk, wkb, D_ * D_);
    nepa_cvt_bf16<<<1024, 256, 0, stream>>>(Wv, wvb, D_ * D_);
    nepa_cvt_bf16<<<1024, 256, 0, stream>>>(Wp, wpb, D_ * D_);

    // 2) QKV projections (WMMA)
    dim3 ggrid(MROWS_ / 128, D_ / 64);
    nepa_gemm_wmma<<<ggrid, 128, 0, stream>>>(xb, wqb, bq, (void*)qf,  0);
    nepa_gemm_wmma<<<ggrid, 128, 0, stream>>>(xb, wkb, bk, (void*)kf,  0);
    nepa_gemm_wmma<<<ggrid, 128, 0, stream>>>(xb, wvb, bv, (void*)vbb, 1);

    // 3) per-head LN + RoPE
    nepa_ln_rope<<<(B_ * H_ * T_) / 8, 256, 0, stream>>>(qf, qn, qbb);
    nepa_ln_rope<<<(B_ * H_ * T_) / 8, 256, 0, stream>>>(kf, kn, kbb);

    // 4) flash attention
    dim3 agrid(T_ / 64, B_ * H_);
    nepa_flash_attn<<<agrid, 128, 0, stream>>>(qbb, kbb, vbb, mask, ctx);

    // 5) output projection -> f32 d_out
    nepa_gemm_wmma<<<ggrid, 128, 0, stream>>>(ctx, wpb, bp, d_out, 2);
}
